// RouterKT_39350490366315
// MI455X (gfx1250) — compile-verified, matrix operand order checked by reading, not run on hardware
//
#include <hip/hip_runtime.h>
#include <hip/hip_bf16.h>
#include <math.h>

// ---------------------------------------------------------------------------
// CDNA5 (gfx1250) RouterKT forward.
// wave32; WMMA f32 <- f16 16x16x32; double-buffered async global->LDS DMA
// staging with partial ASYNCcnt waits (ordered counter pipelining).
// ---------------------------------------------------------------------------

typedef _Float16 v16h __attribute__((ext_vector_type(16)));
typedef float    v8f  __attribute__((ext_vector_type(8)));

#define TILE_M 128
#define TILE_N 64
#define TILE_K 32
#define LDSF   36      // fp32 row stride: 144B, 16B aligned, spreads banks

// CDNA5 async global->LDS copies (ISA 15.18.3 opc 95-98), tracked by ASYNCcnt.
__device__ __forceinline__ void async_copy_b128(unsigned lds_addr, const float* g)
{
    unsigned long long ga = (unsigned long long)(uintptr_t)g;
    asm volatile("global_load_async_to_lds_b128 %0, %1, off"
                 :: "v"(lds_addr), "v"(ga) : "memory");
}
__device__ __forceinline__ void async_copy_b32(unsigned lds_addr, const float* g)
{
    unsigned long long ga = (unsigned long long)(uintptr_t)g;
    asm volatile("global_load_async_to_lds_b32 %0, %1, off"
                 :: "v"(lds_addr), "v"(ga) : "memory");
}

// Flat shared-pointer low 32 bits == wave-relative LDS byte address (ISA 10.2).
__device__ __forceinline__ unsigned lds_addr_of(const void* p)
{
    return (unsigned)(uintptr_t)p;
}

// Generic batched GEMM:
//   C[z][m,n] = epilogue( cscale * (scale_ptr?scale_ptr[z]:1) *
//                         sum_k A[z][m,k] * B[z][n,k] )
// A element: A[offA(z) + m*as_m + k]            (A K-stride is always 1 here)
// B element: B[offB(z) + n*bs_n + k*bs_k]       (bs_k==1 -> NT ; bs_n==1 -> NN)
// z = b*8+h ; off = b*bat?b + h*bat?h
// epilogue: +bias[n], relu, causal mask (mode 1: keep n<=m, mode 2: keep n<m)
__global__ __launch_bounds__(256) void wmma_gemm_kernel(
    const float* __restrict__ A, const float* __restrict__ B,
    const float* __restrict__ bias, float* __restrict__ C,
    int M, int N, int K,
    long as_m, long bs_n, long bs_k, long cs_m,
    long batAb, long batAh, long batBb, long batBh, long batCb, long batCh,
    float cscale, const float* __restrict__ scale_ptr,
    int has_bias, int do_relu, int mask_mode)
{
    __shared__ float As[2][TILE_M][LDSF];   // 2 x 18432 B
    __shared__ float Bs[2][TILE_N][LDSF];   // 2 x  9216 B

    const int tid  = threadIdx.x;
    const int lane = tid & 31;
    const int wave = tid >> 5;       // 8 waves
    const int wm   = wave >> 1;      // 0..3 : 32-row strip
    const int wn   = wave & 1;       // 0..1 : 32-col strip

    const int z  = blockIdx.z;
    const int zb = z >> 3, zh = z & 7;
    const long offA = (long)zb * batAb + (long)zh * batAh;
    const long offB = (long)zb * batBb + (long)zh * batBh;
    const long offC = (long)zb * batCb + (long)zh * batCh;

    const int bm = blockIdx.y * TILE_M;
    const int bn = blockIdx.x * TILE_N;

    const bool nt = (bs_k == 1);

    // ---- Hoisted per-thread staging pointers (advance by constant per step) --
    const float* pA[4];
    unsigned lA[4];
    #pragma unroll
    for (int e = 0; e < 4; ++e) {
        int idx = tid + e * 256, row = idx >> 3, kq = (idx & 7) * 4;
        pA[e] = A + offA + (long)(bm + row) * as_m + kq;
        lA[e] = lds_addr_of(&As[0][row][kq]);
    }
    const float* pB[8];
    unsigned lB[8];
    if (nt) {
        #pragma unroll
        for (int e = 0; e < 2; ++e) {
            int idx = tid + e * 256, row = idx >> 3, kq = (idx & 7) * 4;
            pB[e] = B + offB + (long)(bn + row) * bs_n + kq;
            lB[e] = lds_addr_of(&Bs[0][row][kq]);
        }
    } else {
        #pragma unroll
        for (int e = 0; e < 8; ++e) {
            int idx = tid + e * 256, row = idx >> 5, col = idx & 31;
            pB[e] = B + offB + (long)(bn + row) * bs_n + (long)col * bs_k;
            lB[e] = lds_addr_of(&Bs[0][row][col]);
        }
    }
    const unsigned ABYTES = sizeof(float) * TILE_M * LDSF;   // buffer 1 offset
    const unsigned BBYTES = sizeof(float) * TILE_N * LDSF;
    const long bstep = nt ? (long)TILE_K : (long)TILE_K * bs_k;

    auto stage = [&](int buf) {
        unsigned ao = buf ? ABYTES : 0u;
        unsigned bo = buf ? BBYTES : 0u;
        #pragma unroll
        for (int e = 0; e < 4; ++e) { async_copy_b128(lA[e] + ao, pA[e]); pA[e] += TILE_K; }
        if (nt) {
            #pragma unroll
            for (int e = 0; e < 2; ++e) { async_copy_b128(lB[e] + bo, pB[e]); pB[e] += TILE_K; }
        } else {
            #pragma unroll
            for (int e = 0; e < 8; ++e) { async_copy_b32(lB[e] + bo, pB[e]); pB[e] += bstep; }
        }
    };

    v8f c00 = {}; v8f c01 = {}; v8f c10 = {}; v8f c11 = {};

    const int r16 = lane & 15;
    const int kh8 = (lane >> 4) * 8;   // K-half selector per ISA f16 A/B layout

    // ---- Double-buffered async pipeline ----
    stage(0);
    int cur = 0;
    for (int k0 = 0; k0 < K; k0 += TILE_K) {
        const bool have_next = (k0 + TILE_K) < K;
        if (have_next) stage(cur ^ 1);
        // ASYNCcnt decrements in order: waiting <= (newly issued count)
        // guarantees the *previous* (current-compute) tile group has landed.
        if (!have_next)  asm volatile("s_wait_asynccnt 0x0" ::: "memory");
        else if (nt)     asm volatile("s_wait_asynccnt 0x6" ::: "memory");
        else             asm volatile("s_wait_asynccnt 0xc" ::: "memory");
        __syncthreads();

        const float (*Ac)[LDSF] = As[cur];
        const float (*Bc)[LDSF] = Bs[cur];

        // Fragments (fp32 LDS -> f16) per ISA 7.12.2 16-bit layout.
        v16h a0, a1, b0, b1;
        const int ar0 = wm * 32 + r16;
        const int ar1 = ar0 + 16;
        const int bc0 = wn * 32 + r16;
        const int bc1 = bc0 + 16;
        #pragma unroll
        for (int v = 0; v < 8; ++v) {
            int kb = (v >> 2) * 16 + kh8 + (v & 3) * 2;
            a0[2*v]   = (_Float16)Ac[ar0][kb];  a0[2*v+1] = (_Float16)Ac[ar0][kb+1];
            a1[2*v]   = (_Float16)Ac[ar1][kb];  a1[2*v+1] = (_Float16)Ac[ar1][kb+1];
            b0[2*v]   = (_Float16)Bc[bc0][kb];  b0[2*v+1] = (_Float16)Bc[bc0][kb+1];
            b1[2*v]   = (_Float16)Bc[bc1][kb];  b1[2*v+1] = (_Float16)Bc[bc1][kb+1];
        }
        c00 = __builtin_amdgcn_wmma_f32_16x16x32_f16(false, a0, false, b0, (short)0, c00, false, false);
        c01 = __builtin_amdgcn_wmma_f32_16x16x32_f16(false, a0, false, b1, (short)0, c01, false, false);
        c10 = __builtin_amdgcn_wmma_f32_16x16x32_f16(false, a1, false, b0, (short)0, c10, false, false);
        c11 = __builtin_amdgcn_wmma_f32_16x16x32_f16(false, a1, false, b1, (short)0, c11, false, false);
        __syncthreads();          // all waves done reading 'cur' -> restageable
        cur ^= 1;
    }

    float sc = cscale;
    if (scale_ptr) sc *= scale_ptr[z];

    auto epi = [&](float v, int m, int n) -> float {
        v *= sc;
        if (has_bias) v += bias[n];
        if (do_relu)  v = fmaxf(v, 0.f);
        if (mask_mode == 1)      { if (n > m)  v = -1e9f; }
        else if (mask_mode == 2) { if (n >= m) v = -1e9f; }
        return v;
    };

    // C/D f32 layout: VGPR r -> M = r (lanes 0-15) / r+8 (lanes 16-31), N = lane&15
    const int mb0 = bm + wm * 32 + ((lane >> 4) << 3);
    const int mb1 = mb0 + 16;
    const int n0  = bn + wn * 32 + r16;
    const int n1  = n0 + 16;
    #pragma unroll
    for (int r = 0; r < 8; ++r) {
        int m0 = mb0 + r, m1 = mb1 + r;
        C[offC + (long)m0 * cs_m + n0] = epi(c00[r], m0, n0);
        C[offC + (long)m0 * cs_m + n1] = epi(c01[r], m0, n1);
        C[offC + (long)m1 * cs_m + n0] = epi(c10[r], m1, n0);
        C[offC + (long)m1 * cs_m + n1] = epi(c11[r], m1, n1);
    }
}

// Row softmax over 512 columns, in place. One block per row.
__global__ __launch_bounds__(256) void softmax_kernel(float* __restrict__ S)
{
    __shared__ float red[256];
    float* p = S + (long)blockIdx.x * 512;
    int t = threadIdx.x;
    float a = p[t], b = p[t + 256];
    red[t] = fmaxf(a, b); __syncthreads();
    for (int s = 128; s > 0; s >>= 1) { if (t < s) red[t] = fmaxf(red[t], red[t + s]); __syncthreads(); }
    float mx = red[0]; __syncthreads();
    float ea = __expf(a - mx), eb = __expf(b - mx);
    red[t] = ea + eb; __syncthreads();
    for (int s = 128; s > 0; s >>= 1) { if (t < s) red[t] += red[t + s]; __syncthreads(); }
    float inv = 1.f / red[0];
    p[t] = ea * inv; p[t + 256] = eb * inv;
}

// O[row] = LayerNorm(X[row] + Dl[row]) * g + b   (D = 512, in-place safe)
__global__ __launch_bounds__(256) void add_ln_kernel(
    const float* __restrict__ X, const float* __restrict__ Dl,
    const float* __restrict__ g, const float* __restrict__ b,
    float* __restrict__ O)
{
    __shared__ float red[256];
    long row = blockIdx.x;
    int t = threadIdx.x;
    const float* x = X  + row * 512;
    const float* d = Dl + row * 512;
    float a0 = x[t] + d[t];
    float a1 = x[t + 256] + d[t + 256];
    red[t] = a0 + a1; __syncthreads();
    for (int s = 128; s > 0; s >>= 1) { if (t < s) red[t] += red[t + s]; __syncthreads(); }
    float mu = red[0] * (1.f / 512.f); __syncthreads();
    float d0 = a0 - mu, d1 = a1 - mu;
    red[t] = d0 * d0 + d1 * d1; __syncthreads();
    for (int s = 128; s > 0; s >>= 1) { if (t < s) red[t] += red[t + s]; __syncthreads(); }
    float rstd = rsqrtf(red[0] * (1.f / 512.f) + 1e-5f);
    float* o = O + row * 512;
    o[t]       = d0 * rstd * g[t]       + b[t];
    o[t + 256] = d1 * rstd * g[t + 256] + b[t + 256];
}

// MoH gating: t = q4r.reshape(B,S,H,DK).transpose(0,2,1,3).reshape(B*S,512);
// gates = softmax(t @ wgw^T) over 7, top-2 mask. dyn[r][0]=1, dyn[r][1+j]=kept.
__global__ __launch_bounds__(256) void gate_kernel(
    const float* __restrict__ q4r, const float* __restrict__ wgw, // [7,512]
    float* __restrict__ dyn)                                      // [4096,8]
{
    __shared__ float red[256];
    __shared__ float logit[8];
    int r = blockIdx.x;
    int t = threadIdx.x;
    int cc0 = t, cc1 = t + 256;
    float vals[2];
    int ccs[2] = {cc0, cc1};
    #pragma unroll
    for (int u = 0; u < 2; ++u) {
        long idx = (long)r * 512 + ccs[u];  // flat index into permuted tensor
        int bb = (int)(idx >> 18);
        int hh = (int)((idx >> 15) & 7);
        int ss = (int)((idx >> 6) & 511);
        int dk = (int)(idx & 63);
        vals[u] = q4r[((long)bb * 512 + ss) * 512 + hh * 64 + dk];
    }
    for (int j = 0; j < 7; ++j) {
        float part = vals[0] * wgw[j * 512 + cc0] + vals[1] * wgw[j * 512 + cc1];
        red[t] = part; __syncthreads();
        for (int s = 128; s > 0; s >>= 1) { if (t < s) red[t] += red[t + s]; __syncthreads(); }
        if (t == 0) logit[j] = red[0];
        __syncthreads();
    }
    if (t == 0) {
        float mx = logit[0];
        for (int j = 1; j < 7; ++j) mx = fmaxf(mx, logit[j]);
        float e[7], sum = 0.f;
        for (int j = 0; j < 7; ++j) { e[j] = __expf(logit[j] - mx); sum += e[j]; }
        float inv = 1.f / sum;
        float gg[7];
        for (int j = 0; j < 7; ++j) gg[j] = e[j] * inv;
        int i1 = 0;
        for (int j = 1; j < 7; ++j) if (gg[j] > gg[i1]) i1 = j;
        int i2 = -1;
        for (int j = 0; j < 7; ++j) { if (j == i1) continue; if (i2 < 0 || gg[j] > gg[i2]) i2 = j; }
        dyn[r * 8 + 0] = 1.f;
        for (int j = 0; j < 7; ++j) dyn[r * 8 + 1 + j] = (j == i1 || j == i2) ? gg[j] : 0.f;
    }
}

// rmask[b*8+h] = mean_s dyn[(b*512+s)*8 + h]   (deterministic tree reduce)
__global__ __launch_bounds__(256) void rmask_kernel(const float* __restrict__ dyn,
                                                    float* __restrict__ rmask)
{
    __shared__ float red[256];
    int z = blockIdx.x;            // b*8+h
    int b = z >> 3, h = z & 7;
    int t = threadIdx.x;
    float acc = dyn[((long)(b * 512 + t)) * 8 + h] +
                dyn[((long)(b * 512 + t + 256)) * 8 + h];
    red[t] = acc; __syncthreads();
    for (int s = 128; s > 0; s >>= 1) { if (t < s) red[t] += red[t + s]; __syncthreads(); }
    if (t == 0) rmask[z] = red[0] * (1.f / 512.f);
}

// ---------------------------------------------------------------------------

static inline void launch_gemm(hipStream_t st,
    const float* A, const float* B, const float* bias, float* C,
    int M, int N, int K,
    long as_m, long bs_n, long bs_k, long cs_m,
    long bAb, long bAh, long bBb, long bBh, long bCb, long bCh, int nbat,
    float cscale, const float* sp, int has_bias, int do_relu, int mask_mode)
{
    dim3 grid(N / TILE_N, M / TILE_M, nbat), blk(256);
    hipLaunchKernelGGL(wmma_gemm_kernel, grid, blk, 0, st,
        A, B, bias, C, M, N, K, as_m, bs_n, bs_k, cs_m,
        bAb, bAh, bBb, bBh, bCb, bCh, cscale, sp, has_bias, do_relu, mask_mode);
}

extern "C" void kernel_launch(void* const* d_in, const int* in_sizes, int n_in,
                              void* d_out, int out_size, void* d_ws, size_t ws_size,
                              hipStream_t stream)
{
    (void)in_sizes; (void)n_in; (void)out_size; (void)ws_size;
    const int Bc = 8, Sc = 512, Dc = 512, DFFc = 2048;
    const long MD  = (long)Bc * Sc * Dc;        // 2,097,152 floats
    const long MH  = (long)Bc * Sc * DFFc;      // 8,388,608
    const long SSZ = 64L * Sc * Sc;             // 16,777,216

    const float* q_embed  = (const float*)d_in[0];
    const float* qa_embed = (const float*)d_in[1];
    const float* qw  = (const float*)d_in[2];
    const float* qb  = (const float*)d_in[3];
    const float* vw  = (const float*)d_in[4];
    const float* vb  = (const float*)d_in[5];
    const float* wgw = (const float*)d_in[6];
    const float* ow  = (const float*)d_in[7];
    const float* ob  = (const float*)d_in[8];
    const float* ln1g = (const float*)d_in[9];
    const float* ln1b = (const float*)d_in[10];
    const float* w1  = (const float*)d_in[11];
    const float* b1  = (const float*)d_in[12];
    const float* w2  = (const float*)d_in[13];
    const float* b2  = (const float*)d_in[14];
    const float* ln2g = (const float*)d_in[15];
    const float* ln2b = (const float*)d_in[16];

    float* ws = (float*)d_ws;
    float* x    = ws;            // [B,S,D] q-stream
    float* y    = x + MD;        // [B,S,D] qa-stream
    float* Q    = y + MD;
    float* V    = Q + MD;
    float* O    = V + MD;
    float* T    = O + MD;
    float* Hb   = T + MD;        // FFN hidden [4096,2048]
    float* Sb   = Hb + MH;       // attention [64,512,512]
    float* dyn  = Sb + SSZ;      // [4096,8]
    float* rm   = dyn + 4096L * 8; // [64]

    hipMemcpyAsync(x, q_embed,  MD * sizeof(float), hipMemcpyDeviceToDevice, stream);
    hipMemcpyAsync(y, qa_embed, MD * sizeof(float), hipMemcpyDeviceToDevice, stream);

    const float inv_sqrt_dk = 0.125f;   // 1/sqrt(64)

    auto run_block = [&](int i, int mask_mode, float* query, const float* values,
                         const float* q4r, bool apply_pos)
    {
        const long DD = (long)Dc * Dc;
        const float* qw_i = qw + (long)i * DD;   const float* qb_i = qb + (long)i * Dc;
        const float* vw_i = vw + (long)i * DD;   const float* vb_i = vb + (long)i * Dc;
        const float* wg_i = wgw + (long)i * 7 * Dc;
        const float* ow_i = ow + (long)i * DD;   const float* ob_i = ob + (long)i * Dc;
        const float* g1 = ln1g + (long)i * Dc;   const float* be1 = ln1b + (long)i * Dc;
        const float* g2 = ln2g + (long)i * Dc;   const float* be2 = ln2b + (long)i * Dc;
        const float* w1_i = w1 + (long)i * DFFc * Dc; const float* b1_i = b1 + (long)i * DFFc;
        const float* w2_i = w2 + (long)i * Dc * DFFc; const float* b2_i = b2 + (long)i * Dc;

        // q = query @ qw^T + qb ; v = values @ vw^T + vb
        launch_gemm(stream, query,  qw_i, qb_i, Q, 4096, 512, 512,
                    512, 512, 1, 512, 0,0,0,0,0,0, 1, 1.f, nullptr, 1, 0, 0);
        launch_gemm(stream, values, vw_i, vb_i, V, 4096, 512, 512,
                    512, 512, 1, 512, 0,0,0,0,0,0, 1, 1.f, nullptr, 1, 0, 0);

        // scores[b,h] = Qh @ Qh^T / sqrt(DK), masked.  A/B strided per (b,h).
        launch_gemm(stream, Q, Q, nullptr, Sb, 512, 512, 64,
                    /*as_m*/512, /*bs_n*/512, /*bs_k*/1, /*cs*/512,
                    /*A b,h*/ (long)Sc * Dc, 64, /*B b,h*/ (long)Sc * Dc, 64,
                    /*C b,h*/ 8L * Sc * Sc, (long)Sc * Sc, 64,
                    inv_sqrt_dk, nullptr, 0, 0, mask_mode);

        hipLaunchKernelGGL(softmax_kernel, dim3(64 * 512), dim3(256), 0, stream, Sb);

        // head gating from q4r
        hipLaunchKernelGGL(gate_kernel, dim3(4096), dim3(256), 0, stream, q4r, wg_i, dyn);
        hipLaunchKernelGGL(rmask_kernel, dim3(64), dim3(256), 0, stream, dyn, rm);

        // O[b,s,h*64+d] = (A @ Vh) * rmask[b,h]   (B in NN orientation)
        launch_gemm(stream, Sb, V, nullptr, O, 512, 64, 512,
                    /*as_m*/512, /*bs_n*/1, /*bs_k*/512, /*cs*/512,
                    /*A b,h*/ 8L * Sc * Sc, (long)Sc * Sc,
                    /*B b,h*/ (long)Sc * Dc, 64,
                    /*C b,h*/ (long)Sc * Dc, 64, 64,
                    1.f, rm, 0, 0, 0);

        // o-linear + residual LN
        launch_gemm(stream, O, ow_i, ob_i, T, 4096, 512, 512,
                    512, 512, 1, 512, 0,0,0,0,0,0, 1, 1.f, nullptr, 1, 0, 0);
        hipLaunchKernelGGL(add_ln_kernel, dim3(4096), dim3(256), 0, stream,
                           query, T, g1, be1, query);

        if (apply_pos) {
            launch_gemm(stream, query, w1_i, b1_i, Hb, 4096, 2048, 512,
                        512, 512, 1, 2048, 0,0,0,0,0,0, 1, 1.f, nullptr, 1, 1, 0);
            launch_gemm(stream, Hb, w2_i, b2_i, T, 4096, 512, 2048,
                        2048, 2048, 1, 512, 0,0,0,0,0,0, 1, 1.f, nullptr, 1, 0, 0);
            hipLaunchKernelGGL(add_ln_kernel, dim3(4096), dim3(256), 0, stream,
                               query, T, g2, be2, query);
        }
    };

    // blocks_1: y-stream, causal (j<=i), q4r = original q_embed (x untouched here)
    run_block(0, 1, y, y, x, true);
    run_block(1, 1, y, y, x, true);
    // blocks_2: alternating pairs on x-stream
    run_block(2, 1, x, x, x, false);
    run_block(3, 2, x, y, x, true);
    run_block(4, 1, x, x, x, false);
    run_block(5, 2, x, y, x, true);

    hipMemcpyAsync(d_out, x, MD * sizeof(float), hipMemcpyDeviceToDevice, stream);
}
